// CompactVidTr_31765578121303
// MI455X (gfx1250) — matrix operand order repaired and checked by reading, not compiled, not run on hardware
//
#include <hip/hip_runtime.h>
#include <math.h>

// ---------------- constants ----------------
#define B_    8
#define C_    512
#define F_    2048
#define H_    8
#define HD    64
#define WH    196
#define NCLS  157

typedef __attribute__((ext_vector_type(16))) __bf16 v16bf;
typedef __attribute__((ext_vector_type(8)))  float  v8f;

union F16x {
  v16bf v;
  unsigned short s[16];
  unsigned int   u[8];
};

__device__ __forceinline__ unsigned short f2bf(float f) {
  unsigned int u = __float_as_uint(f);
  unsigned int r = (u + 0x7FFFu + ((u >> 16) & 1u)) >> 16;
  return (unsigned short)r;
}

// pack two floats -> one dword of 2 bf16 (lo = a, hi = b)
__device__ __forceinline__ unsigned int f2bf2(float a, float b) {
  return (unsigned int)f2bf(a) | ((unsigned int)f2bf(b) << 16);
}

__device__ __forceinline__ v8f zero8() {
  v8f z = {0.f,0.f,0.f,0.f,0.f,0.f,0.f,0.f};
  return z;
}

__device__ __forceinline__ v8f wmma_bf(const F16x& a, const F16x& b, v8f c) {
  return __builtin_amdgcn_wmma_f32_16x16x32_bf16(false, a.v, false, b.v,
                                                 (short)0, c, false, false);
}

// ---------------- generic bf16 WMMA GEMM ----------------
// C[M,N] = A[M,K] @ B[K,N] (+bias[N]) (+res[M,N]) (relu)
// TRANS_B: B stored [N,K] row-major (weight^T access).
// Requires: K % 32 == 0, N % 128 == 0, 16B-aligned A/B bases.
template<bool TRANS_B, bool RELU, bool HAS_RES>
__global__ __launch_bounds__(256) void gemm_bf16(
    const float* __restrict__ A, const float* __restrict__ Bm,
    const float* __restrict__ bias, const float* __restrict__ res,
    float* __restrict__ Cout, int M, int N, int K)
{
  __shared__ unsigned short As[128][34];   // [m][k] bf16, pad keeps dword align (68B rows)
  __shared__ unsigned short Bs[128][34];   // [n][k] bf16
  const int tid  = threadIdx.x;
  const int lane = tid & 31;
  const int wv   = tid >> 5;
  const int lidx = lane & 15;
  const int hb   = (lane >> 4) & 1;        // 0: lanes0-15, 1: lanes16-31
  const int bm = blockIdx.y * 128;
  const int bn = blockIdx.x * 128;
  const int wm = (wv >> 1) * 32;
  const int wn = (wv & 1) * 64;

  v8f acc[2][4];
  for (int r = 0; r < 2; ++r)
    for (int c = 0; c < 4; ++c) acc[r][c] = zero8();

  const int kTiles = K >> 5;
  for (int kt = 0; kt < kTiles; ++kt) {
    const int k0 = kt << 5;
    // A tile: 128x32 fp32 -> bf16, 1024 float4 loads
    for (int vid = tid; vid < 1024; vid += 256) {
      int r = vid >> 3, c4 = (vid & 7) << 2;
      int gr = bm + r;
      float4 val = make_float4(0.f, 0.f, 0.f, 0.f);
      if (gr < M) val = *(const float4*)(A + (size_t)gr * K + k0 + c4);
      As[r][c4 + 0] = f2bf(val.x);
      As[r][c4 + 1] = f2bf(val.y);
      As[r][c4 + 2] = f2bf(val.z);
      As[r][c4 + 3] = f2bf(val.w);
    }
    // B tile: -> Bs[n][k]
    if (TRANS_B) {
      for (int vid = tid; vid < 1024; vid += 256) {
        int n = vid >> 3, c4 = (vid & 7) << 2;
        int gn = bn + n;
        float4 val = *(const float4*)(Bm + (size_t)gn * K + k0 + c4);
        Bs[n][c4 + 0] = f2bf(val.x);
        Bs[n][c4 + 1] = f2bf(val.y);
        Bs[n][c4 + 2] = f2bf(val.z);
        Bs[n][c4 + 3] = f2bf(val.w);
      }
    } else {
      for (int vid = tid; vid < 1024; vid += 256) {
        int kk = vid >> 5, n4 = (vid & 31) << 2;
        int gk = k0 + kk;
        float4 val = *(const float4*)(Bm + (size_t)gk * N + bn + n4);
        Bs[n4 + 0][kk] = f2bf(val.x);
        Bs[n4 + 1][kk] = f2bf(val.y);
        Bs[n4 + 2][kk] = f2bf(val.z);
        Bs[n4 + 3][kk] = f2bf(val.w);
      }
    }
    __syncthreads();

    F16x afr[2], bfr[4];
    for (int rt = 0; rt < 2; ++rt) {
      int m  = wm + rt * 16 + lidx;
      int kb = hb ? 8 : 0;
#pragma unroll
      for (int i = 0; i < 8; ++i) {
        int kk = ((i < 4) ? 2 * i : 16 + 2 * (i - 4)) + kb;   // even
        afr[rt].u[i] = *(const unsigned int*)&As[m][kk];
      }
    }
    for (int ct = 0; ct < 4; ++ct) {
      int n  = wn + ct * 16 + lidx;
      int kb = hb ? 8 : 0;
#pragma unroll
      for (int i = 0; i < 8; ++i) {
        int kk = ((i < 4) ? 2 * i : 16 + 2 * (i - 4)) + kb;   // even
        bfr[ct].u[i] = *(const unsigned int*)&Bs[n][kk];
      }
    }
    for (int rt = 0; rt < 2; ++rt)
      for (int ct = 0; ct < 4; ++ct)
        acc[rt][ct] = wmma_bf(afr[rt], bfr[ct], acc[rt][ct]);
    __syncthreads();
  }

  for (int rt = 0; rt < 2; ++rt)
    for (int ct = 0; ct < 4; ++ct) {
      int n = bn + wn + ct * 16 + lidx;
      int mbase = bm + wm + rt * 16 + (hb ? 8 : 0);
      for (int j = 0; j < 8; ++j) {
        int m = mbase + j;
        if (m < M) {
          float v = acc[rt][ct][j] + bias[n];
          if (HAS_RES) v += res[(size_t)m * N + n];
          if (RELU)    v = fmaxf(v, 0.f);
          Cout[(size_t)m * N + n] = v;
        }
      }
    }
}

// ---------------- im2col for the 16x16/stride16 patch conv ----------------
__global__ __launch_bounds__(256) void im2col_kernel(const float* __restrict__ src,
                                                     float* __restrict__ A)
{
  const size_t total4 = (size_t)25088 * 192;   // rows x (768/4) float4s
  for (size_t id = (size_t)blockIdx.x * 256 + threadIdx.x; id < total4;
       id += (size_t)gridDim.x * 256) {
    int kq = (int)(id % 192);
    size_t row = id / 192;
    int b = (int)(row % B_);
    size_t tok = row / B_;
    int wx = (int)(tok % 14);
    int hy = (int)((tok / 14) % 14);
    int ti = (int)(tok / 196);
    int k  = kq << 2;          // 768-dim index, px-multiple-of-4
    int px = k & 15;
    int py = (k >> 4) & 15;
    int ch = k >> 8;
    float4 v = *(const float4*)(src +
        (((size_t)(b * 3 + ch) * 16 + ti) * 224 + (hy * 16 + py)) * 224
        + (wx * 16 + px));
    *(float4*)(A + id * 4) = v;
  }
}

__global__ __launch_bounds__(256) void fill_cls_kernel(const float* __restrict__ cls,
                                                       float* __restrict__ x)
{
  int id = blockIdx.x * 256 + threadIdx.x;
  if (id < B_ * C_) x[id] = cls[id % C_];   // x[0,b,c] = cls_tok[c]
}

// ---------------- layernorm over C=512 per (token,batch) row ----------------
__global__ __launch_bounds__(256) void layernorm_kernel(
    const float* __restrict__ xin, const float* __restrict__ w,
    const float* __restrict__ bb, const float* __restrict__ pos,
    float* __restrict__ out)
{
  const int row = blockIdx.x;
  const int tid = threadIdx.x;
  const float* xr = xin + (size_t)row * C_;
  float v0 = xr[tid], v1 = xr[tid + 256];
  __shared__ float red[256];
  red[tid] = v0 + v1;
  __syncthreads();
  for (int s = 128; s > 0; s >>= 1) { if (tid < s) red[tid] += red[tid + s]; __syncthreads(); }
  float mean = red[0] * (1.f / 512.f);
  __syncthreads();
  float d0 = v0 - mean, d1 = v1 - mean;
  red[tid] = d0 * d0 + d1 * d1;
  __syncthreads();
  for (int s = 128; s > 0; s >>= 1) { if (tid < s) red[tid] += red[tid + s]; __syncthreads(); }
  float rstd = rsqrtf(red[0] * (1.f / 512.f) + 1e-5f);
  float o0 = d0 * rstd * w[tid]       + bb[tid];
  float o1 = d1 * rstd * w[tid + 256] + bb[tid + 256];
  if (pos) {
    int l = row / B_;
    o0 += pos[(size_t)l * C_ + tid];
    o1 += pos[(size_t)l * C_ + tid + 256];
  }
  out[(size_t)row * C_ + tid]       = o0;
  out[(size_t)row * C_ + tid + 256] = o1;
}

// ---------------- cls-token attention (q = token 0, keys = all L) ----------------
__global__ __launch_bounds__(256) void cls_attn_kernel(const float* __restrict__ qkv,
                                                       float* __restrict__ att, int L)
{
  const int b = blockIdx.x >> 3;
  const int h = blockIdx.x & 7;
  const int tid = threadIdx.x;
  __shared__ float q[HD];
  __shared__ float sc[3137];
  __shared__ float red[256];
  if (tid < HD) q[tid] = qkv[(size_t)b * 1536 + h * HD + tid] * 0.125f;
  __syncthreads();
  float lmax = -1e30f;
  for (int l = tid; l < L; l += 256) {
    const float* kp = qkv + ((size_t)l * B_ + b) * 1536 + 512 + h * HD;
    float d = 0.f;
    for (int c = 0; c < HD; c += 4) {
      float4 kv = *(const float4*)(kp + c);
      d += q[c] * kv.x + q[c + 1] * kv.y + q[c + 2] * kv.z + q[c + 3] * kv.w;
    }
    sc[l] = d;
    lmax = fmaxf(lmax, d);
  }
  red[tid] = lmax; __syncthreads();
  for (int s = 128; s > 0; s >>= 1) { if (tid < s) red[tid] = fmaxf(red[tid], red[tid + s]); __syncthreads(); }
  float m = red[0]; __syncthreads();
  float lsum = 0.f;
  for (int l = tid; l < L; l += 256) { float e = __expf(sc[l] - m); sc[l] = e; lsum += e; }
  red[tid] = lsum; __syncthreads();
  for (int s = 128; s > 0; s >>= 1) { if (tid < s) red[tid] += red[tid + s]; __syncthreads(); }
  float inv = 1.f / red[0];
  __syncthreads();
  const int c = tid & 63, g = tid >> 6;
  float a = 0.f;
  for (int l = g; l < L; l += 4)
    a += sc[l] * qkv[((size_t)l * B_ + b) * 1536 + 1024 + h * HD + c];
  red[tid] = a; __syncthreads();
  if (g == 0) {
    float s = red[c] + red[64 + c] + red[128 + c] + red[192 + c];
    att[(size_t)b * C_ + h * HD + c] = s * inv;
  }
}

// -------- temporal attention per (h,b,w): t<=16 queries/keys, WMMA + sigma top-k --------
__global__ __launch_bounds__(32) void temporal_attn_kernel(
    const float* __restrict__ qkv, float* __restrict__ y, int* __restrict__ idxg,
    int t, int pk, int tp)
{
  const int bid = blockIdx.x;
  const int w = bid % WH;
  const int b = (bid / WH) % B_;
  const int h = bid / (WH * B_);
  const int lane = threadIdx.x;
  const int hb = (lane >> 4) & 1;
  const int lidx = lane & 15;
  const size_t RS = 1536;
  const size_t hoff = (size_t)h * HD;

  F16x qa[2];
  {
    int m = lidx; bool mv = m < t;
    size_t qrow = mv ? ((size_t)(1 + m * WH + w) * B_ + b) * RS + hoff : 0;
    for (int f = 0; f < 2; ++f)
#pragma unroll
      for (int i = 0; i < 8; ++i) {
        int d = f * 32 + ((i < 4) ? 2 * i : 16 + 2 * (i - 4)) + (hb ? 8 : 0);
        float2 p = mv ? *(const float2*)(qkv + qrow + d) : make_float2(0.f, 0.f);
        qa[f].u[i] = f2bf2(p.x * 0.125f, p.y * 0.125f);
      }
  }
  F16x kb[2];
  {
    int n = lidx; bool nv = n < t;
    size_t krow = nv ? ((size_t)(1 + n * WH + w) * B_ + b) * RS + 512 + hoff : 0;
    for (int f = 0; f < 2; ++f)
#pragma unroll
      for (int i = 0; i < 8; ++i) {
        int d = f * 32 + ((i < 4) ? 2 * i : 16 + 2 * (i - 4)) + (hb ? 8 : 0);
        float2 p = nv ? *(const float2*)(qkv + krow + d) : make_float2(0.f, 0.f);
        kb[f].u[i] = f2bf2(p.x, p.y);
      }
  }
  v8f s = zero8();
  s = wmma_bf(qa[0], kb[0], s);
  s = wmma_bf(qa[1], kb[1], s);

  const bool cv = lidx < t;
  const float invt = 1.f / (float)t;
  __shared__ float sAt[16][18];   // 72B rows: float2-aligned
  __shared__ float sSig[16];
  __shared__ int   sIdx[16];
  for (int j = 0; j < 8; ++j) {
    float sv = cv ? s[j] : -1e30f;
    float mx = sv;
    for (int off = 1; off < 16; off <<= 1) mx = fmaxf(mx, __shfl_xor(mx, off));
    float e = cv ? __expf(sv - mx) : 0.f;
    float sum = e;
    for (int off = 1; off < 16; off <<= 1) sum += __shfl_xor(sum, off);
    float pr = e / sum;
    float sq = pr * pr;
    for (int off = 1; off < 16; off <<= 1) sq += __shfl_xor(sq, off);
    int r = j + hb * 8;
    sAt[r][lidx] = pr;
    if (lidx == 0) {
      float var = (t > 1) ? (sq - invt) / (float)(t - 1) : 0.f;
      sSig[r] = sqrtf(fmaxf(var, 0.f));
    }
  }
  __syncthreads();
  if (lane == 0) {
    bool drop[16];
    for (int q2 = 0; q2 < 16; ++q2) drop[q2] = false;
    for (int d = 0; d < pk; ++d) {        // drop pk smallest-sigma rows (tie: later idx)
      float best = 1e30f; int bi = 0;
      for (int q2 = 0; q2 < t; ++q2)
        if (!drop[q2] && sSig[q2] <= best) { best = sSig[q2]; bi = q2; }
      drop[bi] = true;
    }
    int c2 = 0;
    for (int q2 = 0; q2 < t; ++q2) if (!drop[q2]) sIdx[c2++] = q2;
  }
  __syncthreads();
  if (pk > 0 && lane < tp)
    idxg[((size_t)(h * B_ + b) * WH + w) * tp + lane] = sIdx[lane];

  F16x pa;
  {
    int m = lidx; bool mv = m < tp;
    int sr = mv ? sIdx[m] : 0;
#pragma unroll
    for (int i = 0; i < 8; ++i) {
      int k = ((i < 4) ? 2 * i : 16 + 2 * (i - 4)) + (hb ? 8 : 0);   // even
      float2 p = *(const float2*)&sAt[sr][k];
      float x0 = (mv && k     < t) ? p.x : 0.f;
      float x1 = (mv && k + 1 < t) ? p.y : 0.f;
      pa.u[i] = f2bf2(x0, x1);
    }
  }
  for (int dt = 0; dt < 4; ++dt) {
    F16x vb;
#pragma unroll
    for (int i = 0; i < 8; ++i) {
      int k = ((i < 4) ? 2 * i : 16 + 2 * (i - 4)) + (hb ? 8 : 0);
      float x0 = (k < t) ?
        qkv[((size_t)(1 + k * WH + w) * B_ + b) * RS + 1024 + hoff + dt * 16 + lidx] : 0.f;
      float x1 = (k + 1 < t) ?
        qkv[((size_t)(1 + (k + 1) * WH + w) * B_ + b) * RS + 1024 + hoff + dt * 16 + lidx] : 0.f;
      vb.u[i] = f2bf2(x0, x1);
    }
    v8f o = wmma_bf(pa, vb, zero8());
    for (int j = 0; j < 8; ++j) {
      int m = j + hb * 8;
      if (m < tp)
        y[((size_t)(m * WH + w) * B_ + b) * C_ + hoff + dt * 16 + lidx] = o[j];
    }
  }
}

// -------- spatial attention per (h,b,ti,qtile): 196 keys, flash softmax + WMMA --------
__global__ __launch_bounds__(32) void spatial_attn_kernel(
    const float* __restrict__ qkvs, float* __restrict__ att, int tp)
{
  const int bid = blockIdx.x;
  const int qt = bid % 13;
  const int ti = (bid / 13) % tp;
  const int b  = (bid / (13 * tp)) % B_;
  const int h  = bid / (13 * tp * B_);
  const int lane = threadIdx.x;
  const int hb = (lane >> 4) & 1;
  const int lidx = lane & 15;
  const size_t RS = 1536;
  const size_t hoff = (size_t)h * HD;

  F16x qa[2];
  {
    int m = qt * 16 + lidx; bool mv = m < WH;
    size_t qrow = mv ? ((size_t)(ti * WH + m) * B_ + b) * RS + hoff : 0;
    for (int f = 0; f < 2; ++f)
#pragma unroll
      for (int i = 0; i < 8; ++i) {
        int d = f * 32 + ((i < 4) ? 2 * i : 16 + 2 * (i - 4)) + (hb ? 8 : 0);
        float2 p = mv ? *(const float2*)(qkvs + qrow + d) : make_float2(0.f, 0.f);
        qa[f].u[i] = f2bf2(p.x * 0.125f, p.y * 0.125f);
      }
  }
  float mrun[8], ssum[8];
  for (int j = 0; j < 8; ++j) { mrun[j] = -1e30f; ssum[j] = 0.f; }
  v8f o[4];
  for (int dt = 0; dt < 4; ++dt) o[dt] = zero8();
  __shared__ float sP[16][34];    // 136B rows: float2-aligned

  for (int kc = 0; kc < 7; ++kc) {            // 7 chunks of 32 keys (196 keys)
    const int kb0 = kc * 32;
    v8f st[2];
    for (int kt2 = 0; kt2 < 2; ++kt2) {
      int key = kb0 + kt2 * 16 + lidx;
      bool kvv = key < WH;
      size_t krow = kvv ? ((size_t)(ti * WH + key) * B_ + b) * RS + 512 + hoff : 0;
      F16x kf[2];
      for (int f = 0; f < 2; ++f)
#pragma unroll
        for (int i = 0; i < 8; ++i) {
          int d = f * 32 + ((i < 4) ? 2 * i : 16 + 2 * (i - 4)) + (hb ? 8 : 0);
          float2 p = kvv ? *(const float2*)(qkvs + krow + d) : make_float2(0.f, 0.f);
          kf[f].u[i] = f2bf2(p.x, p.y);
        }
      v8f z = zero8();
      z = wmma_bf(qa[0], kf[0], z);
      z = wmma_bf(qa[1], kf[1], z);
      st[kt2] = z;
    }
    const bool c0v = (kb0 + lidx) < WH;
    const bool c1v = (kb0 + 16 + lidx) < WH;
    for (int j = 0; j < 8; ++j) {
      float s0 = c0v ? st[0][j] : -1e30f;
      float s1 = c1v ? st[1][j] : -1e30f;
      float lm = fmaxf(s0, s1);
      for (int off = 1; off < 16; off <<= 1) lm = fmaxf(lm, __shfl_xor(lm, off));
      float nm = fmaxf(mrun[j], lm);
      float sc = __expf(mrun[j] - nm);
      float p0 = c0v ? __expf(s0 - nm) : 0.f;
      float p1 = c1v ? __expf(s1 - nm) : 0.f;
      float ps = p0 + p1;
      for (int off = 1; off < 16; off <<= 1) ps += __shfl_xor(ps, off);
      ssum[j] = ssum[j] * sc + ps;
      mrun[j] = nm;
      for (int dt = 0; dt < 4; ++dt) o[dt][j] *= sc;
      int r = j + hb * 8;
      sP[r][lidx] = p0;
      sP[r][16 + lidx] = p1;
    }
    __syncthreads();
    F16x pa;
#pragma unroll
    for (int i = 0; i < 8; ++i) {
      int k = ((i < 4) ? 2 * i : 16 + 2 * (i - 4)) + (hb ? 8 : 0);   // even
      float2 p = *(const float2*)&sP[lidx][k];
      pa.u[i] = f2bf2(p.x, p.y);
    }
    __syncthreads();
    for (int dt = 0; dt < 4; ++dt) {
      F16x vf;
#pragma unroll
      for (int i = 0; i < 8; ++i) {
        int k = kb0 + ((i < 4) ? 2 * i : 16 + 2 * (i - 4)) + (hb ? 8 : 0);
        float x0 = (k < WH) ?
          qkvs[((size_t)(ti * WH + k) * B_ + b) * RS + 1024 + hoff + dt * 16 + lidx] : 0.f;
        float x1 = (k + 1 < WH) ?
          qkvs[((size_t)(ti * WH + (k + 1)) * B_ + b) * RS + 1024 + hoff + dt * 16 + lidx] : 0.f;
        vf.u[i] = f2bf2(x0, x1);
      }
      o[dt] = wmma_bf(pa, vf, o[dt]);
    }
  }
  for (int dt = 0; dt < 4; ++dt)
    for (int j = 0; j < 8; ++j) {
      int q = qt * 16 + j + hb * 8;
      if (q < WH && ssum[j] > 0.f)
        att[((size_t)(1 + ti * WH + q) * B_ + b) * C_ + hoff + dt * 16 + lidx] =
            o[dt][j] / ssum[j];
    }
}

// ---------------- pooled residual gather ----------------
__global__ __launch_bounds__(256) void pool_residual_kernel(
    const float* __restrict__ xin, const int* __restrict__ idxg,
    float* __restrict__ xout, int tp)
{
  const size_t total = (size_t)(1 + tp * WH) * B_ * C_;
  for (size_t id = (size_t)blockIdx.x * 256 + threadIdx.x; id < total;
       id += (size_t)gridDim.x * 256) {
    int c = (int)(id % C_);
    int b = (int)((id / C_) % B_);
    size_t l = id / ((size_t)C_ * B_);
    if (l == 0) {
      xout[id] = xin[(size_t)b * C_ + c];
    } else {
      size_t lm = l - 1;
      int w = (int)(lm % WH);
      int m = (int)(lm / WH);
      int h = c / HD;
      int sti = idxg[((size_t)(h * B_ + b) * WH + w) * tp + m];
      xout[id] = xin[((size_t)(1 + sti * WH + w) * B_ + b) * C_ + c];
    }
  }
}

// ---------------- classifier head ----------------
__global__ __launch_bounds__(256) void head_kernel(
    const float* __restrict__ xln, const float* __restrict__ fcw,
    const float* __restrict__ fcb, float* __restrict__ out)
{
  int id = blockIdx.x * 256 + threadIdx.x;
  if (id >= B_ * NCLS) return;
  int n = id % NCLS, b = id / NCLS;
  float a = fcb[n];
  for (int c = 0; c < C_; ++c)
    a += xln[(size_t)b * C_ + c] * fcw[(size_t)c * NCLS + n];
  out[id] = a;
}

// ---------------- workspace layout (float units) ----------------
static const size_t SZX      = (size_t)3137 * B_ * C_;        // x buffer (max L)
static const size_t OFF_X0   = 0;
static const size_t OFF_X1   = SZX;
static const size_t OFF_LN   = 2 * SZX;
static const size_t OFF_ATT  = 3 * SZX;
static const size_t OFF_Y    = 4 * SZX;
static const size_t OFF_BIG  = 5 * SZX;                       // im2col / qkv_t / ffn-h
static const size_t BIGSZ    = (size_t)3137 * B_ * 2048;
static const size_t OFF_QKVS = OFF_BIG + BIGSZ;
static const size_t QKVSSZ   = (size_t)16 * WH * B_ * 1536;
static const size_t OFF_IDX  = OFF_QKVS + QKVSSZ;             // ints

extern "C" void kernel_launch(void* const* d_in, const int* in_sizes, int n_in,
                              void* d_out, int out_size, void* d_ws, size_t ws_size,
                              hipStream_t stream) {
  (void)in_sizes; (void)n_in; (void)out_size; (void)ws_size;
  const float* src   = (const float*)d_in[0];
  const float* convw = (const float*)d_in[1];
  const float* convb = (const float*)d_in[2];
  const float* cls   = (const float*)d_in[3];
  const float* pos   = (const float*)d_in[4];
  const float* Wt    = (const float*)d_in[5];
  const float* bt    = (const float*)d_in[6];
  const float* Ws    = (const float*)d_in[7];
  const float* bs    = (const float*)d_in[8];
  const float* Wo    = (const float*)d_in[9];
  const float* bo    = (const float*)d_in[10];
  const float* n1w   = (const float*)d_in[11];
  const float* n1b   = (const float*)d_in[12];
  const float* n2w   = (const float*)d_in[13];
  const float* n2b   = (const float*)d_in[14];
  const float* W1    = (const float*)d_in[15];
  const float* b1    = (const float*)d_in[16];
  const float* W2    = (const float*)d_in[17];
  const float* b2    = (const float*)d_in[18];
  const float* enw   = (const float*)d_in[19];
  const float* enb   = (const float*)d_in[20];
  const float* fcw   = (const float*)d_in[21];
  const float* fcb   = (const float*)d_in[22];

  float* wsf  = (float*)d_ws;
  float* X0   = wsf + OFF_X0;
  float* X1   = wsf + OFF_X1;
  float* LNB  = wsf + OFF_LN;
  float* ATT  = wsf + OFF_ATT;
  float* Y    = wsf + OFF_Y;
  float* BIG  = wsf + OFF_BIG;
  float* QKVS = wsf + OFF_QKVS;
  int*   IDX  = (int*)(wsf + OFF_IDX);

  // patch embed: im2col + WMMA GEMM (A[25088,768] @ conv_w^T[768,512]) + bias
  im2col_kernel<<<4096, 256, 0, stream>>>(src, BIG);
  {
    dim3 g(512 / 128, (25088 + 127) / 128);
    gemm_bf16<true, false, false><<<g, 256, 0, stream>>>(
        BIG, convw, convb, nullptr, X0 + (size_t)B_ * C_, 25088, 512, 768);
  }
  fill_cls_kernel<<<(B_ * C_ + 255) / 256, 256, 0, stream>>>(cls, X0);

  float* xc = X0;
  float* xo = X1;
  int t = 16;
  const int pks[6] = {0, 2, 0, 2, 0, 2};
  for (int i = 0; i < 6; ++i) {
    const int L = 1 + t * WH;
    const int M = L * B_;
    layernorm_kernel<<<M, 256, 0, stream>>>(
        xc, n1w + i * C_, n1b + i * C_, (i == 0) ? pos : nullptr, LNB);
    {
      dim3 g(1536 / 128, (M + 127) / 128);
      gemm_bf16<false, false, false><<<g, 256, 0, stream>>>(
          LNB, Wt + (size_t)i * C_ * 1536, bt + i * 1536, nullptr, BIG, M, 1536, C_);
    }
    const int pk = pks[i];
    const int tp = t - pk;
    cls_attn_kernel<<<B_ * H_, 256, 0, stream>>>(BIG, ATT, L);
    temporal_attn_kernel<<<H_ * B_ * WH, 32, 0, stream>>>(BIG, Y, IDX, t, pk, tp);
    const int Ms = tp * WH * B_;
    {
      dim3 g(1536 / 128, (Ms + 127) / 128);
      gemm_bf16<false, false, false><<<g, 256, 0, stream>>>(
          Y, Ws + (size_t)i * C_ * 1536, bs + i * 1536, nullptr, QKVS, Ms, 1536, C_);
    }
    spatial_attn_kernel<<<H_ * B_ * tp * 13, 32, 0, stream>>>(QKVS, ATT, tp);
    const int Lo = 1 + tp * WH;
    if (pk > 0) {
      pool_residual_kernel<<<2048, 256, 0, stream>>>(xc, IDX, xo, tp);
      float* tmp = xc; xc = xo; xo = tmp;
    }
    const int Mo = Lo * B_;
    {
      dim3 g(C_ / 128, (Mo + 127) / 128);
      gemm_bf16<false, false, true><<<g, 256, 0, stream>>>(
          ATT, Wo + (size_t)i * C_ * C_, bo + i * C_, xc, xc, Mo, C_, C_);
    }
    layernorm_kernel<<<Mo, 256, 0, stream>>>(
        xc, n2w + i * C_, n2b + i * C_, nullptr, LNB);
    {
      dim3 g(F_ / 128, (Mo + 127) / 128);
      gemm_bf16<false, true, false><<<g, 256, 0, stream>>>(
          LNB, W1 + (size_t)i * C_ * F_, b1 + i * F_, nullptr, BIG, Mo, F_, C_);
    }
    {
      dim3 g(C_ / 128, (Mo + 127) / 128);
      gemm_bf16<false, false, true><<<g, 256, 0, stream>>>(
          BIG, W2 + (size_t)i * F_ * C_, b2 + i * C_, xc, xc, Mo, C_, F_);
    }
    t = tp;
  }
  layernorm_kernel<<<B_, 256, 0, stream>>>(xc, enw, enb, nullptr, LNB);
  head_kernel<<<(B_ * NCLS + 255) / 256, 256, 0, stream>>>(LNB, fcw, fcb, (float*)d_out);
}
